// FreqMlp_43791486550365
// MI455X (gfx1250) — compile-verified, matrix operand order checked by reading, not run on hardware
//
#include <hip/hip_runtime.h>
#include <hip/hip_bf16.h>

// ---------------- problem dims ----------------
#define B_   128
#define F_   256
#define C_   512
#define H_   2048
#define M_   (B_ * (F_ / 2))   // 16384 rows of (b, fr)
#define CH_  (C_ / 2)          // 256

typedef __attribute__((ext_vector_type(16))) __bf16 v16bf;
typedef __attribute__((ext_vector_type(8)))  float  v8f;

union FragBF16 {
    v16bf v;
    uint4 q[2];   // q[0] = elements 0..7 (16B), q[1] = elements 8..15 (16B)
};

__device__ __forceinline__ unsigned short f32_to_bf16_rne(float f) {
    unsigned int u = __float_as_uint(f);
    u += 0x7FFFu + ((u >> 16) & 1u);   // round-to-nearest-even
    return (unsigned short)(u >> 16);
}

// ---------------------------------------------------------------
// Kernel 1: Haar LL band + linear upsample 256 -> 512, output bf16
// ll_up is (M_, C_) row-major, k-contiguous (WMMA A operand friendly)
// ---------------------------------------------------------------
__global__ void k_llup(const float* __restrict__ x, unsigned short* __restrict__ llup) {
    int u = blockIdx.x * blockDim.x + threadIdx.x;
    if (u >= M_ * C_) return;
    int m  = u >> 9;          // / 512
    int o  = u & 511;
    int b  = m >> 7;
    int fr = m & 127;
    float pos = (float)o * (255.0f / 511.0f);
    int   i0  = (int)pos;
    if (i0 > 254) i0 = 254;
    float w = pos - (float)i0;
    const float* xp = x + ((size_t)(b * F_ + 2 * fr) * C_);
    int j0 = 2 * i0;
    float a0 = xp[j0],     b0 = xp[j0 + 1],     c0 = xp[j0 + C_],     d0 = xp[j0 + C_ + 1];
    float a1 = xp[j0 + 2], b1 = xp[j0 + 3],     c1 = xp[j0 + C_ + 2], d1 = xp[j0 + C_ + 3];
    float ll0 = 0.5f * (a0 + b0 + c0 + d0);
    float ll1 = 0.5f * (a1 + b1 + c1 + d1);
    llup[u] = f32_to_bf16_rne(ll0 + w * (ll1 - ll0));
}

// ---------------------------------------------------------------
// Kernel 2: convert both weight matrices to bf16 (L2-resident afterwards)
// ---------------------------------------------------------------
__global__ void k_cvt(const float* __restrict__ w1, const float* __restrict__ w2,
                      unsigned short* __restrict__ w1b, unsigned short* __restrict__ w2b) {
    int u  = blockIdx.x * blockDim.x + threadIdx.x;
    const int n1 = H_ * C_;
    const int n2 = CH_ * H_;
    if (u < n1) {
        w1b[u] = f32_to_bf16_rne(w1[u]);
    } else {
        int u2 = u - n1;
        if (u2 < n2) w2b[u2] = f32_to_bf16_rne(w2[u2]);
    }
}

// ---------------------------------------------------------------
// Software-pipelined 32Mx64N wave mainloop.
// Pointers are pre-offset by the lane's K-half (koff); each k-step of 32
// advances by 4 uint4 (64B). Fragment chunk 2 lives at +2 uint4 (k+16).
// ---------------------------------------------------------------
__device__ __forceinline__ void wmma_mainloop_32x64(
    const uint4* __restrict__ ap0, const uint4* __restrict__ ap1,
    const uint4* __restrict__ bp0, const uint4* __restrict__ bp1,
    const uint4* __restrict__ bp2, const uint4* __restrict__ bp3,
    const int ksteps, v8f acc[2][4])
{
    FragBF16 a0, a1, b0, b1, b2, b3;
    a0.q[0] = ap0[0]; a0.q[1] = ap0[2];
    a1.q[0] = ap1[0]; a1.q[1] = ap1[2];
    b0.q[0] = bp0[0]; b0.q[1] = bp0[2];
    b1.q[0] = bp1[0]; b1.q[1] = bp1[2];
    b2.q[0] = bp2[0]; b2.q[1] = bp2[2];
    b3.q[0] = bp3[0]; b3.q[1] = bp3[2];

    for (int s = 1; s < ksteps; ++s) {
        const int q = 4 * s;
        // issue next-stage loads into fresh registers (partial loadcnt waits)
        FragBF16 na0, na1, nb0, nb1, nb2, nb3;
        na0.q[0] = ap0[q]; na0.q[1] = ap0[q + 2];
        na1.q[0] = ap1[q]; na1.q[1] = ap1[q + 2];
        nb0.q[0] = bp0[q]; nb0.q[1] = bp0[q + 2];
        nb1.q[0] = bp1[q]; nb1.q[1] = bp1[q + 2];
        nb2.q[0] = bp2[q]; nb2.q[1] = bp2[q + 2];
        nb3.q[0] = bp3[q]; nb3.q[1] = bp3[q + 2];
        __builtin_prefetch(&ap0[q + 4], 0, 3);
        __builtin_prefetch(&ap1[q + 4], 0, 3);
        // 8 WMMAs on current stage while next-stage loads are in flight
        acc[0][0] = __builtin_amdgcn_wmma_f32_16x16x32_bf16(false, a0.v, false, b0.v, (short)0, acc[0][0], false, false);
        acc[0][1] = __builtin_amdgcn_wmma_f32_16x16x32_bf16(false, a0.v, false, b1.v, (short)0, acc[0][1], false, false);
        acc[0][2] = __builtin_amdgcn_wmma_f32_16x16x32_bf16(false, a0.v, false, b2.v, (short)0, acc[0][2], false, false);
        acc[0][3] = __builtin_amdgcn_wmma_f32_16x16x32_bf16(false, a0.v, false, b3.v, (short)0, acc[0][3], false, false);
        acc[1][0] = __builtin_amdgcn_wmma_f32_16x16x32_bf16(false, a1.v, false, b0.v, (short)0, acc[1][0], false, false);
        acc[1][1] = __builtin_amdgcn_wmma_f32_16x16x32_bf16(false, a1.v, false, b1.v, (short)0, acc[1][1], false, false);
        acc[1][2] = __builtin_amdgcn_wmma_f32_16x16x32_bf16(false, a1.v, false, b2.v, (short)0, acc[1][2], false, false);
        acc[1][3] = __builtin_amdgcn_wmma_f32_16x16x32_bf16(false, a1.v, false, b3.v, (short)0, acc[1][3], false, false);
        a0 = na0; a1 = na1; b0 = nb0; b1 = nb1; b2 = nb2; b3 = nb3;
    }
    // epilogue stage
    acc[0][0] = __builtin_amdgcn_wmma_f32_16x16x32_bf16(false, a0.v, false, b0.v, (short)0, acc[0][0], false, false);
    acc[0][1] = __builtin_amdgcn_wmma_f32_16x16x32_bf16(false, a0.v, false, b1.v, (short)0, acc[0][1], false, false);
    acc[0][2] = __builtin_amdgcn_wmma_f32_16x16x32_bf16(false, a0.v, false, b2.v, (short)0, acc[0][2], false, false);
    acc[0][3] = __builtin_amdgcn_wmma_f32_16x16x32_bf16(false, a0.v, false, b3.v, (short)0, acc[0][3], false, false);
    acc[1][0] = __builtin_amdgcn_wmma_f32_16x16x32_bf16(false, a1.v, false, b0.v, (short)0, acc[1][0], false, false);
    acc[1][1] = __builtin_amdgcn_wmma_f32_16x16x32_bf16(false, a1.v, false, b1.v, (short)0, acc[1][1], false, false);
    acc[1][2] = __builtin_amdgcn_wmma_f32_16x16x32_bf16(false, a1.v, false, b2.v, (short)0, acc[1][2], false, false);
    acc[1][3] = __builtin_amdgcn_wmma_f32_16x16x32_bf16(false, a1.v, false, b3.v, (short)0, acc[1][3], false, false);
}

// ---------------------------------------------------------------
// GEMM1: h = gelu(ll_up @ W1^T + b1), output bf16 (M_ x H_)
// block = 8 waves as 4Mx2N -> block tile 128M x 128N; wave tile 32M x 64N
// ---------------------------------------------------------------
__global__ __launch_bounds__(256) void k_gemm1(
    const unsigned short* __restrict__ A,    // M_ x C_   bf16
    const unsigned short* __restrict__ Bm,   // H_ x C_   bf16 (n rows, k contiguous)
    const float* __restrict__ bias,          // H_
    unsigned short* __restrict__ Hout)       // M_ x H_   bf16
{
    const int lane  = threadIdx.x & 31;
    const int wv    = threadIdx.x >> 5;
    const int row16 = lane & 15;
    const int qoff  = (lane >> 4);           // K-half per lane group, in uint4 units
    const int mBase = blockIdx.y * 128 + (wv & 3) * 32;
    const int nBase = blockIdx.x * 128 + (wv >> 2) * 64;

    const uint4* ap0 = (const uint4*)(A  + (size_t)(mBase +  0 + row16) * C_) + qoff;
    const uint4* ap1 = (const uint4*)(A  + (size_t)(mBase + 16 + row16) * C_) + qoff;
    const uint4* bp0 = (const uint4*)(Bm + (size_t)(nBase +  0 + row16) * C_) + qoff;
    const uint4* bp1 = (const uint4*)(Bm + (size_t)(nBase + 16 + row16) * C_) + qoff;
    const uint4* bp2 = (const uint4*)(Bm + (size_t)(nBase + 32 + row16) * C_) + qoff;
    const uint4* bp3 = (const uint4*)(Bm + (size_t)(nBase + 48 + row16) * C_) + qoff;

    v8f acc[2][4] = {{v8f{}, v8f{}, v8f{}, v8f{}}, {v8f{}, v8f{}, v8f{}, v8f{}}};
    wmma_mainloop_32x64(ap0, ap1, bp0, bp1, bp2, bp3, C_ / 32, acc);

    // epilogue: bias + exact GELU, store bf16
    const int mOff = (lane >> 4) << 3;        // C/D rows: lanes16-31 -> M+8
    #pragma unroll
    for (int mt = 0; mt < 2; ++mt) {
        #pragma unroll
        for (int t = 0; t < 4; ++t) {
            const int n = nBase + t * 16 + (lane & 15);
            const float bv = bias[n];
            #pragma unroll
            for (int r = 0; r < 8; ++r) {
                const int m = mBase + mt * 16 + mOff + r;
                float v = acc[mt][t][r] + bv;
                float g = 0.5f * v * (1.0f + erff(v * 0.70710678118654752f));
                Hout[(size_t)m * H_ + n] = f32_to_bf16_rne(g);
            }
        }
    }
}

// ---------------------------------------------------------------
// GEMM2: ll_new = h @ W2^T + b2, fused with inverse Haar DWT epilogue
// block tile 128M x 128N; wave tile 32M x 64N
// ---------------------------------------------------------------
__global__ __launch_bounds__(256) void k_gemm2(
    const unsigned short* __restrict__ Hin,  // M_ x H_  bf16
    const unsigned short* __restrict__ Bm,   // CH_ x H_ bf16
    const float* __restrict__ bias,          // CH_
    const float* __restrict__ x,             // B_ x F_ x C_ (for LH/HL/HH bands)
    float* __restrict__ out)                 // B_ x F_ x C_
{
    const int lane  = threadIdx.x & 31;
    const int wv    = threadIdx.x >> 5;
    const int row16 = lane & 15;
    const int qoff  = (lane >> 4);
    const int mBase = blockIdx.y * 128 + (wv & 3) * 32;
    const int nBase = blockIdx.x * 128 + (wv >> 2) * 64;

    const uint4* ap0 = (const uint4*)(Hin + (size_t)(mBase +  0 + row16) * H_) + qoff;
    const uint4* ap1 = (const uint4*)(Hin + (size_t)(mBase + 16 + row16) * H_) + qoff;
    const uint4* bp0 = (const uint4*)(Bm  + (size_t)(nBase +  0 + row16) * H_) + qoff;
    const uint4* bp1 = (const uint4*)(Bm  + (size_t)(nBase + 16 + row16) * H_) + qoff;
    const uint4* bp2 = (const uint4*)(Bm  + (size_t)(nBase + 32 + row16) * H_) + qoff;
    const uint4* bp3 = (const uint4*)(Bm  + (size_t)(nBase + 48 + row16) * H_) + qoff;

    v8f acc[2][4] = {{v8f{}, v8f{}, v8f{}, v8f{}}, {v8f{}, v8f{}, v8f{}, v8f{}}};
    wmma_mainloop_32x64(ap0, ap1, bp0, bp1, bp2, bp3, H_ / 32, acc);

    // epilogue: bias, recompute LH/HL/HH bands from x, inverse Haar, store out
    const int mOff = (lane >> 4) << 3;
    #pragma unroll
    for (int mt = 0; mt < 2; ++mt) {
        #pragma unroll
        for (int t = 0; t < 4; ++t) {
            const int o = nBase + t * 16 + (lane & 15);   // 0..255
            const float bv = bias[o];
            #pragma unroll
            for (int r = 0; r < 8; ++r) {
                const int m  = mBase + mt * 16 + mOff + r;
                const int b  = m >> 7;
                const int fr = m & 127;
                const size_t base = ((size_t)(b * F_ + 2 * fr) * C_) + 2 * o;
                const float xa = x[base],      xb = x[base + 1];
                const float xc = x[base + C_], xd = x[base + C_ + 1];
                const float lh = 0.5f * (xa + xb - xc - xd);
                const float hl = 0.5f * (xa - xb + xc - xd);
                const float hh = 0.5f * (xa - xb - xc + xd);
                const float ll = acc[mt][t][r] + bv;
                out[base]          = 0.5f * (ll + lh + hl + hh);
                out[base + 1]      = 0.5f * (ll + lh - hl - hh);
                out[base + C_]     = 0.5f * (ll - lh + hl - hh);
                out[base + C_ + 1] = 0.5f * (ll - lh - hl + hh);
            }
        }
    }
}

// ---------------------------------------------------------------
extern "C" void kernel_launch(void* const* d_in, const int* in_sizes, int n_in,
                              void* d_out, int out_size, void* d_ws, size_t ws_size,
                              hipStream_t stream) {
    const float* x     = (const float*)d_in[0];
    const float* fc1_w = (const float*)d_in[1];
    const float* fc1_b = (const float*)d_in[2];
    const float* fc2_w = (const float*)d_in[3];
    const float* fc2_b = (const float*)d_in[4];
    float* out = (float*)d_out;

    // workspace layout (bf16 buffers): ll_up (16MB) | W1 (2MB) | W2 (1MB) | h (64MB)
    char* ws = (char*)d_ws;
    unsigned short* llup = (unsigned short*)ws;
    unsigned short* w1b  = (unsigned short*)(ws + (size_t)M_ * C_ * 2);
    unsigned short* w2b  = (unsigned short*)(ws + (size_t)M_ * C_ * 2 + (size_t)H_ * C_ * 2);
    unsigned short* hbuf = (unsigned short*)(ws + (size_t)M_ * C_ * 2 + (size_t)H_ * C_ * 2 +
                                                  (size_t)CH_ * H_ * 2);

    {   // DWT-LL + upsample -> bf16
        const int total = M_ * C_;
        k_llup<<<(total + 255) / 256, 256, 0, stream>>>(x, llup);
    }
    {   // weight conversion -> bf16
        const int total = H_ * C_ + CH_ * H_;
        k_cvt<<<(total + 255) / 256, 256, 0, stream>>>(fc1_w, fc2_w, w1b, w2b);
    }
    // GEMM1 + GELU: (16384 x 512) @ (512 x 2048), block tile 128x128
    k_gemm1<<<dim3(H_ / 128, M_ / 128), 256, 0, stream>>>(llup, w1b, fc1_b, hbuf);
    // GEMM2 + inverse DWT: (16384 x 2048) @ (2048 x 256) -> out, block tile 128x128
    k_gemm2<<<dim3(CH_ / 128, M_ / 128), 256, 0, stream>>>(hbuf, w2b, fc2_b, x, out);
}
